// AttentionBlock_81612968558608
// MI455X (gfx1250) — compile-verified, matrix operand order checked by reading, not run on hardware
//
#include <hip/hip_runtime.h>
#include <hip/hip_bf16.h>
#include <hip/hip_fp16.h>

typedef __attribute__((ext_vector_type(16))) _Float16 v16h;
typedef __attribute__((ext_vector_type(8)))  _Float16 v8h;
typedef __attribute__((ext_vector_type(4)))  _Float16 v4h;
typedef __attribute__((ext_vector_type(8)))  float    v8f;

#define NB     16
#define CC     512
#define TT     1024
#define NHEADS 8
#define DHEAD  64
#define GROUPS 32

// D = A*B + C, f16 inputs, f32 accumulate, 16x16x32
__device__ __forceinline__ v8f wmma16(v16h a, v16h b, v8f c) {
    return __builtin_amdgcn_wmma_f32_16x16x32_f16(
        false, a, false, b, (short)0, c, false, false);
}

// A/B fragment loader: lane-half lh (0 = lanes 0-15, 1 = lanes 16-31).
// halves 0..7  -> k = lh*8 + 0..7 ; halves 8..15 -> k = 16 + lh*8 + 0..7
__device__ __forceinline__ v16h load_frag(const _Float16* row, int lh) {
    v16h f;
    const int b0 = lh * 8, b1 = 16 + lh * 8;
#pragma unroll
    for (int i = 0; i < 8; ++i) { f[i] = row[b0 + i]; f[8 + i] = row[b1 + i]; }
    return f;
}

// CDNA5 async copy: 16 bytes global -> LDS, tracked by ASYNCcnt (no VGPR data).
__device__ __forceinline__ void async_copy_b128(unsigned lds_off, const void* gptr) {
    asm volatile("global_load_async_to_lds_b128 %0, %1, off"
                 :: "v"(lds_off), "v"(gptr) : "memory");
}
__device__ __forceinline__ void wait_async0() {
    asm volatile("s_wait_asynccnt 0" ::: "memory");
}

// ---------------- one-shot weight f32 -> f16 conversion --------------------
__global__ void __launch_bounds__(256) cvt_kernel(const float* __restrict__ wq,
                                                  const float* __restrict__ wp,
                                                  _Float16* __restrict__ oq,
                                                  _Float16* __restrict__ op) {
    const int idx = blockIdx.x * 256 + threadIdx.x;
    const int stride = gridDim.x * 256;
    const int nq4 = (3 * CC * CC) / 4, np4 = (CC * CC) / 4;
    for (int i = idx; i < nq4; i += stride) {
        const float4 v = ((const float4*)wq)[i];
        v4h h; h[0] = (_Float16)v.x; h[1] = (_Float16)v.y; h[2] = (_Float16)v.z; h[3] = (_Float16)v.w;
        ((v4h*)oq)[i] = h;
    }
    for (int i = idx; i < np4; i += stride) {
        const float4 v = ((const float4*)wp)[i];
        v4h h; h[0] = (_Float16)v.x; h[1] = (_Float16)v.y; h[2] = (_Float16)v.z; h[3] = (_Float16)v.w;
        ((v4h*)op)[i] = h;
    }
}

// ---------------- GroupNorm: per (b, group) block; writes xnT[b][t][c] -----
__global__ void __launch_bounds__(256) gn_kernel(const float* __restrict__ x,
                                                 const float* __restrict__ gsc,
                                                 const float* __restrict__ gbi,
                                                 _Float16* __restrict__ xnT) {
    const int blk = blockIdx.x;          // b*GROUPS + g
    const int b = blk >> 5, g = blk & 31;
    const int CPG = CC / GROUPS;         // 16 channels per group
    const int N = CPG * TT;              // 16384 elements
    const float* xp = x + ((size_t)b * CC + (size_t)g * CPG) * TT;
    const int tid = threadIdx.x;

    float s = 0.f, s2 = 0.f;
    for (int i = tid; i < N; i += 256) { float v = xp[i]; s += v; s2 += v * v; }

    __shared__ float red0[256], red1[256];
    red0[tid] = s; red1[tid] = s2;
    __syncthreads();
    for (int off = 128; off > 0; off >>= 1) {
        if (tid < off) { red0[tid] += red0[tid + off]; red1[tid] += red1[tid + off]; }
        __syncthreads();
    }
    const float mu   = red0[0] / (float)N;
    const float var  = red1[0] / (float)N - mu * mu;
    const float rinv = rsqrtf(var + 1e-5f);

    _Float16* op = xnT + (size_t)b * TT * CC;   // [t][c], 2B scatters merged in L2
    for (int i = tid; i < N; i += 256) {
        const int c = g * CPG + (i >> 10);      // i / TT
        const int t = i & (TT - 1);
        const float v = (xp[i] - mu) * rinv * gsc[c] + gbi[c];
        op[(size_t)t * CC + c] = (_Float16)v;
    }
}

// ---------------- QKV GEMM -> attention-native q/k/v layouts ---------------
__global__ void __launch_bounds__(256, 1) qkv_kernel(const _Float16* __restrict__ xnT,
                                                     const _Float16* __restrict__ wq,
                                                     const float* __restrict__ bias,
                                                     _Float16* __restrict__ qT,   // [b][h][t][d]
                                                     _Float16* __restrict__ kT,   // [b][h][s][d]
                                                     _Float16* __restrict__ vv) { // [b][h][d][s]
    // grid (T/128, 3C/128, B), 256 threads = 8 waves; wave tile 32(O) x 64(T)
    const int b  = blockIdx.z;
    const int o0 = blockIdx.y * 128;
    const int t0 = blockIdx.x * 128;
    const int tid = threadIdx.x;
    const int wave = tid >> 5, lane = tid & 31;
    const int lh = lane >> 4, ln = lane & 15;
    const int wy = wave >> 1, wx = wave & 1;   // 4x2 wave grid

    __shared__ alignas(16) _Float16 wl[2][128][40];   // 80B rows: B128-aligned
    __shared__ alignas(16) _Float16 xl[2][128][40];

    v8f acc[2][4] = {};
    const _Float16* xb = xnT + (size_t)b * TT * CC;

    auto stage = [&](int buf, int c0) {
#pragma unroll
        for (int u = 0; u < 2; ++u) {                // W rows: 128 x 4 chunks
            const int i = tid + u * 256;
            const int r = i >> 2, ch = (i & 3) * 8;
            async_copy_b128((unsigned)(size_t)&wl[buf][r][ch],
                            wq + (size_t)(o0 + r) * CC + c0 + ch);
        }
#pragma unroll
        for (int u = 0; u < 2; ++u) {                // X rows: 128 x 4 chunks
            const int i = tid + u * 256;
            const int t = i >> 2, ch = (i & 3) * 8;
            async_copy_b128((unsigned)(size_t)&xl[buf][t][ch],
                            xb + (size_t)(t0 + t) * CC + c0 + ch);
        }
    };

    stage(0, 0);
    wait_async0();
    __syncthreads();

    int buf = 0;
    for (int k = 0; k < CC / 32; ++k) {
        if (k + 1 < CC / 32) stage(buf ^ 1, (k + 1) * 32);   // overlap with compute
        v16h af[2], bf[4];
#pragma unroll
        for (int mi = 0; mi < 2; ++mi) af[mi] = load_frag(&wl[buf][wy * 32 + mi * 16 + ln][0], lh);
#pragma unroll
        for (int j = 0; j < 4; ++j)    bf[j]  = load_frag(&xl[buf][wx * 64 + j * 16 + ln][0], lh);
#pragma unroll
        for (int mi = 0; mi < 2; ++mi)
#pragma unroll
            for (int j = 0; j < 4; ++j)
                acc[mi][j] = wmma16(af[mi], bf[j], acc[mi][j]);
        wait_async0();
        __syncthreads();
        buf ^= 1;
    }
    // epilogue: o splits as (typ, head, d); typ is uniform per block (128 | 512)
#pragma unroll
    for (int mi = 0; mi < 2; ++mi)
#pragma unroll
    for (int j = 0; j < 4; ++j) {
        const int t  = t0 + wx * 64 + j * 16 + ln;
        const int ob = o0 + wy * 32 + mi * 16 + lh * 8;      // 8-aligned
        const int typ = ob >> 9, hh = (ob >> 6) & 7, dd0 = ob & 63;
        if (typ < 2) {
            v8h hv;
#pragma unroll
            for (int r = 0; r < 8; ++r) hv[r] = (_Float16)(acc[mi][j][r] + bias[ob + r]);
            _Float16* dst = (typ == 0 ? qT : kT) +
                            ((size_t)(b * NHEADS + hh) * TT + t) * DHEAD + dd0;
            *(v8h*)dst = hv;                                  // packed 16B store
        } else {
#pragma unroll
            for (int r = 0; r < 8; ++r) {
                const float v = acc[mi][j][r] + bias[ob + r];
                vv[((size_t)(b * NHEADS + hh) * DHEAD + dd0 + r) * TT + t] = (_Float16)v;
            }
        }
    }
}

// ---------------- Flash attention: all tiles via async copy ----------------
__global__ void __launch_bounds__(128, 1) attn_kernel(const _Float16* __restrict__ qT,
                                                      const _Float16* __restrict__ kT,
                                                      const _Float16* __restrict__ vv,
                                                      _Float16* __restrict__ aT) {
    // grid (T/64, NHEADS, B), 128 threads = 4 waves; 64-key double-buffered blocks
    const int b = blockIdx.z, h = blockIdx.y;
    const int tq0 = blockIdx.x * 64;
    const int tid = threadIdx.x;
    const int wave = tid >> 5, lane = tid & 31;
    const int lh = lane >> 4, ln = lane & 15;

    const _Float16* qb = qT + (size_t)(b * NHEADS + h) * TT * DHEAD;  // [t][d]
    const _Float16* kb = kT + (size_t)(b * NHEADS + h) * TT * DHEAD;  // [s][d]
    const _Float16* vb = vv + (size_t)(b * NHEADS + h) * DHEAD * TT;  // [d][s]

    __shared__ alignas(16) _Float16 qt[64][72];      // [t][d]
    __shared__ alignas(16) _Float16 kt[2][64][72];   // [s][d]
    __shared__ alignas(16) _Float16 vt[2][64][72];   // [d][s]
    __shared__ _Float16 pt[4][16][66];               // per-wave P tile [row][s]

#pragma unroll
    for (int u = 0; u < 4; ++u) {                    // Q tile: pure row copy
        const int i = tid + u * 128;
        const int t = i >> 3, ch = (i & 7) * 8;
        async_copy_b128((unsigned)(size_t)&qt[t][ch], qb + (size_t)(tq0 + t) * DHEAD + ch);
    }

    auto stageKV = [&](int bufn, int s0) {
#pragma unroll
        for (int u = 0; u < 4; ++u) {                // K tile: row copy
            const int i = tid + u * 128;
            const int s = i >> 3, ch = (i & 7) * 8;
            async_copy_b128((unsigned)(size_t)&kt[bufn][s][ch],
                            kb + (size_t)(s0 + s) * DHEAD + ch);
        }
#pragma unroll
        for (int u = 0; u < 4; ++u) {                // V tile: row copy
            const int i = tid + u * 128;
            const int dd = i >> 3, ch = (i & 7) * 8;
            async_copy_b128((unsigned)(size_t)&vt[bufn][dd][ch],
                            vb + (size_t)dd * TT + s0 + ch);
        }
    };

    stageKV(0, 0);
    wait_async0();
    __syncthreads();

    const int qrow = wave * 16 + ln;
    v16h qa[2];
    qa[0] = load_frag(&qt[qrow][0], lh);
    qa[1] = load_frag(&qt[qrow][32], lh);

    float m[8], l[8];
#pragma unroll
    for (int r = 0; r < 8; ++r) { m[r] = -__builtin_inff(); l[r] = 0.f; }
    v8f oacc[4] = {};

    int buf = 0;
    for (int it = 0; it < TT / 64; ++it) {
        if (it + 1 < TT / 64) stageKV(buf ^ 1, (it + 1) * 64);  // overlap with compute

        // scores: 16 rows x 64 keys (1/sqrt(d) folded in afterwards in f32)
        v8f st[4] = {};
#pragma unroll
        for (int j = 0; j < 4; ++j) {
#pragma unroll
            for (int ch = 0; ch < 2; ++ch) {
                const v16h bf = load_frag(&kt[buf][j * 16 + ln][ch * 32], lh);
                st[j] = wmma16(qa[ch], bf, st[j]);
            }
        }

        // online softmax (row r of this lane-half lives in one VGPR slot)
        float corr[8];
#pragma unroll
        for (int r = 0; r < 8; ++r) {
            float mx = -__builtin_inff();
#pragma unroll
            for (int j = 0; j < 4; ++j) {
                st[j][r] *= 0.125f;
                mx = fmaxf(mx, st[j][r]);
            }
#pragma unroll
            for (int d = 1; d < 16; d <<= 1) mx = fmaxf(mx, __shfl_xor(mx, d, 32));
            const float mn = fmaxf(m[r], mx);
            corr[r] = __expf(m[r] - mn);
            m[r] = mn;
            float rs = 0.f;
#pragma unroll
            for (int j = 0; j < 4; ++j) {
                const float p = __expf(st[j][r] - mn);
                st[j][r] = p;
                rs += p;
            }
#pragma unroll
            for (int d = 1; d < 16; d <<= 1) rs += __shfl_xor(rs, d, 32);
            l[r] = l[r] * corr[r] + rs;
        }
#pragma unroll
        for (int j = 0; j < 4; ++j)
#pragma unroll
            for (int r = 0; r < 8; ++r) oacc[j][r] *= corr[r];

        // D-layout -> A-layout via LDS
#pragma unroll
        for (int r = 0; r < 8; ++r) {
            const int row = r + lh * 8;
#pragma unroll
            for (int j = 0; j < 4; ++j)
                pt[wave][row][j * 16 + ln] = (_Float16)st[j][r];
        }
        __syncthreads();

        v16h pa[2];
        pa[0] = load_frag(&pt[wave][ln][0], lh);
        pa[1] = load_frag(&pt[wave][ln][32], lh);
#pragma unroll
        for (int j = 0; j < 4; ++j) {
#pragma unroll
            for (int ch = 0; ch < 2; ++ch) {
                const v16h bf = load_frag(&vt[buf][j * 16 + ln][ch * 32], lh);
                oacc[j] = wmma16(pa[ch], bf, oacc[j]);
            }
        }
        wait_async0();
        __syncthreads();
        buf ^= 1;
    }

    // epilogue: normalize, store transposed aT[b][t][h*64+dd] (L2 merges scatters)
    _Float16* ap = aT + (size_t)b * TT * CC + (size_t)h * DHEAD;
#pragma unroll
    for (int j = 0; j < 4; ++j) {
        const int dd = j * 16 + ln;
#pragma unroll
        for (int r = 0; r < 8; ++r) {
            const int t = tq0 + wave * 16 + r + lh * 8;
            const float v = oacc[j][r] / l[r];
            ap[(size_t)t * CC + dd] = (_Float16)v;
        }
    }
}

// ---------------- Proj GEMM + bias + residual, async double-buffered -------
__global__ void __launch_bounds__(256, 1) proj_kernel(const _Float16* __restrict__ aT,
                                                      const _Float16* __restrict__ wp,
                                                      const float* __restrict__ bias,
                                                      const float* __restrict__ x,
                                                      float* __restrict__ out) {
    // grid (T/128, C/128, B), 256 threads = 8 waves
    const int b  = blockIdx.z;
    const int o0 = blockIdx.y * 128;
    const int t0 = blockIdx.x * 128;
    const int tid = threadIdx.x;
    const int wave = tid >> 5, lane = tid & 31;
    const int lh = lane >> 4, ln = lane & 15;
    const int wy = wave >> 1, wx = wave & 1;

    __shared__ alignas(16) _Float16 wl[2][128][40];
    __shared__ alignas(16) _Float16 xl[2][128][40];

    v8f acc[2][4] = {};
    const _Float16* ab = aT + (size_t)b * TT * CC;

    auto stage = [&](int buf, int c0) {
#pragma unroll
        for (int u = 0; u < 2; ++u) {
            const int i = tid + u * 256;
            const int r = i >> 2, ch = (i & 3) * 8;
            async_copy_b128((unsigned)(size_t)&wl[buf][r][ch],
                            wp + (size_t)(o0 + r) * CC + c0 + ch);
        }
#pragma unroll
        for (int u = 0; u < 2; ++u) {
            const int i = tid + u * 256;
            const int t = i >> 2, ch = (i & 3) * 8;
            async_copy_b128((unsigned)(size_t)&xl[buf][t][ch],
                            ab + (size_t)(t0 + t) * CC + c0 + ch);
        }
    };

    stage(0, 0);
    wait_async0();
    __syncthreads();

    int buf = 0;
    for (int k = 0; k < CC / 32; ++k) {
        if (k + 1 < CC / 32) stage(buf ^ 1, (k + 1) * 32);
        v16h af[2], bf[4];
#pragma unroll
        for (int mi = 0; mi < 2; ++mi) af[mi] = load_frag(&wl[buf][wy * 32 + mi * 16 + ln][0], lh);
#pragma unroll
        for (int j = 0; j < 4; ++j)    bf[j]  = load_frag(&xl[buf][wx * 64 + j * 16 + ln][0], lh);
#pragma unroll
        for (int mi = 0; mi < 2; ++mi)
#pragma unroll
            for (int j = 0; j < 4; ++j)
                acc[mi][j] = wmma16(af[mi], bf[j], acc[mi][j]);
        wait_async0();
        __syncthreads();
        buf ^= 1;
    }
#pragma unroll
    for (int mi = 0; mi < 2; ++mi)
#pragma unroll
    for (int j = 0; j < 4; ++j) {
        const int t = t0 + wx * 64 + j * 16 + ln;
#pragma unroll
        for (int r = 0; r < 8; ++r) {
            const int o = o0 + wy * 32 + mi * 16 + r + lh * 8;
            const size_t idx = ((size_t)b * CC + o) * TT + t;
            out[idx] = x[idx] + bias[o] + acc[mi][j][r];
        }
    }
}

extern "C" void kernel_launch(void* const* d_in, const int* in_sizes, int n_in,
                              void* d_out, int out_size, void* d_ws, size_t ws_size,
                              hipStream_t stream) {
    (void)in_sizes; (void)n_in; (void)out_size; (void)ws_size;
    const float* x    = (const float*)d_in[0];
    const float* gsc  = (const float*)d_in[1];
    const float* gbi  = (const float*)d_in[2];
    const float* qkvw = (const float*)d_in[3];
    const float* qkvb = (const float*)d_in[4];
    const float* pw   = (const float*)d_in[5];
    const float* pb   = (const float*)d_in[6];
    float* out = (float*)d_out;

    char* ws = (char*)d_ws;
    _Float16* xnT  = (_Float16*)(ws);                              // 16 MB  [b][t][c]
    _Float16* qT   = (_Float16*)(ws + (size_t)16 * 1024 * 1024);   // 16 MB  [b][h][t][d]
    _Float16* kT   = (_Float16*)(ws + (size_t)32 * 1024 * 1024);   // 16 MB  [b][h][s][d]
    _Float16* vv   = (_Float16*)(ws + (size_t)48 * 1024 * 1024);   // 16 MB  [b][h][d][s]
    _Float16* aT   = (_Float16*)(ws + (size_t)64 * 1024 * 1024);   // 16 MB  [b][t][c]
    _Float16* wq16 = (_Float16*)(ws + (size_t)80 * 1024 * 1024);   // 1.5 MB
    _Float16* wp16 = (_Float16*)(ws + (size_t)82 * 1024 * 1024);   // 0.5 MB

    cvt_kernel <<<dim3(512), 256, 0, stream>>>(qkvw, pw, wq16, wp16);
    gn_kernel  <<<dim3(NB * GROUPS), 256, 0, stream>>>(x, gsc, gbi, xnT);
    qkv_kernel <<<dim3(TT / 128, (3 * CC) / 128, NB), 256, 0, stream>>>(xnT, wq16, qkvb, qT, kT, vv);
    attn_kernel<<<dim3(TT / 64, NHEADS, NB), 128, 0, stream>>>(qT, kT, vv, aT);
    proj_kernel<<<dim3(TT / 128, CC / 128, NB), 256, 0, stream>>>(aT, wp16, pb, x, out);
}
